// PGNNLayer_5634997092467
// MI455X (gfx1250) — compile-verified, matrix operand order checked by reading, not run on hardware
//
#include <hip/hip_runtime.h>
#include <hip/hip_bf16.h>

typedef __attribute__((ext_vector_type(16))) __bf16 v16bf;
typedef __attribute__((ext_vector_type(8)))  __bf16 v8bf;
typedef __attribute__((ext_vector_type(8)))  float  v8f;

#define NNODES   50000
#define KANCH    64
#define IN_DIM   256
#define OUT_DIM  64

#define TILES_PER_BLOCK 4
#define NTILES   ((NNODES + 15) / 16)                                  // 3125
#define NBLK_A   ((NTILES + TILES_PER_BLOCK - 1) / TILES_PER_BLOCK)    // 782
#define KPAD     264   // row stride in bf16: 264*2B = 528B = 132 dw == 4 banks (mod 64)

// ---------------------------------------------------------------------------
// Kernel A: u_feat = X@W_u + b_u ; v_feat = X@W_v + b_v
// Block = 256 threads = 8 waves; wave w -> matrix (w>>2), col-tile (w&3)*16.
// Both weight matrices are staged bf16-transposed in LDS once per block, and
// 4 row-tiles (64 nodes) are processed per block to amortize the staging.
// A/B fragments are then pure ds_load_b128 traffic; WMMA: bf16 16x16x32.
// ---------------------------------------------------------------------------
__global__ __launch_bounds__(256)
void pgnn_proj_wmma(const float* __restrict__ feature,
                    const float* __restrict__ Wu, const float* __restrict__ bu,
                    const float* __restrict__ Wv, const float* __restrict__ bv,
                    float* __restrict__ u_feat, float* __restrict__ v_feat)
{
    __shared__ __bf16 sW[2][OUT_DIM][KPAD];              // [mat][col][k] (transposed)
    __shared__ __bf16 sX[TILES_PER_BLOCK * 16][KPAD];    // [row][k]

    const int t = threadIdx.x;

    // ---- stage weights: coalesced global b32, transposed bf16 into LDS ----
    #pragma unroll
    for (int m = 0; m < 2; ++m) {
        const float* __restrict__ W = m ? Wv : Wu;
        for (int idx = t; idx < IN_DIM * OUT_DIM; idx += 256) {
            const int k = idx >> 6, col = idx & 63;      // W is [k][col] row-major
            sW[m][col][k] = (__bf16)W[idx];
        }
    }

    // ---- stage this block's 64 feature rows as bf16 (float4 global loads) ----
    const int row_base = blockIdx.x * (TILES_PER_BLOCK * 16);
    for (int idx4 = t; idx4 < TILES_PER_BLOCK * 16 * (IN_DIM / 4); idx4 += 256) {
        const int row = idx4 / (IN_DIM / 4);
        const int k4  = (idx4 % (IN_DIM / 4)) * 4;
        float4 f = make_float4(0.f, 0.f, 0.f, 0.f);
        if (row_base + row < NNODES)
            f = *(const float4*)(feature + (long)(row_base + row) * IN_DIM + k4);
        __bf16* p = &sX[row][k4];
        p[0] = (__bf16)f.x; p[1] = (__bf16)f.y; p[2] = (__bf16)f.z; p[3] = (__bf16)f.w;
    }
    __syncthreads();

    const int lane  = threadIdx.x & 31;
    const int wv    = threadIdx.x >> 5;
    const int mat   = wv >> 2;            // 0 = W_u path, 1 = W_v path
    const int n0    = (wv & 3) * 16;      // output column tile base
    const int col   = n0 + (lane & 15);
    const int khalf = (lane >> 4) * 8;    // K sub-offset per half-wave

    const float* __restrict__ bias = mat ? bv : bu;
    float*       __restrict__ outp = mat ? v_feat : u_feat;
    const float  bcol = bias[col];

    for (int tt = 0; tt < TILES_PER_BLOCK; ++tt) {
        const int tile = blockIdx.x * TILES_PER_BLOCK + tt;
        if (tile >= NTILES) break;                 // wave-uniform guard
        const int m0   = tile * 16;
        const int srow = tt * 16 + (lane & 15);

        v8f c = {};
        #pragma unroll
        for (int ks = 0; ks < IN_DIM / 32; ++ks) {
            const int klo = ks * 32 + khalf;
            const v8bf alo = *(const v8bf*)&sX[srow][klo];
            const v8bf ahi = *(const v8bf*)&sX[srow][klo + 16];
            const v8bf blo = *(const v8bf*)&sW[mat][col][klo];
            const v8bf bhi = *(const v8bf*)&sW[mat][col][klo + 16];
            const v16bf a = __builtin_shufflevector(alo, ahi,
                0,1,2,3,4,5,6,7,8,9,10,11,12,13,14,15);
            const v16bf b = __builtin_shufflevector(blo, bhi,
                0,1,2,3,4,5,6,7,8,9,10,11,12,13,14,15);
            c = __builtin_amdgcn_wmma_f32_16x16x32_bf16(
                    /*neg_a=*/false, a, /*neg_b=*/false, b,
                    /*c_mod=*/(short)0, c, /*reuse_a=*/false, /*reuse_b=*/false);
        }

        // D layout: VGPR r = row m0+r (lanes 0-15) / m0+r+8 (lanes 16-31)
        #pragma unroll
        for (int r = 0; r < 8; ++r) {
            const int orow = m0 + r + khalf;
            outp[(long)orow * OUT_DIM + col] = c[r] + bcol;
        }
    }
}

// ---------------------------------------------------------------------------
// Kernel B: fused edge/anchor phase. One wave per node; lane owns channels
// {2*lane, 2*lane+1}. u_feat/v_feat (25.6 MB) stay resident in the 192 MB L2,
// so the 1.6 GB of per-anchor gathers are L2 hits, never HBM.
// ---------------------------------------------------------------------------
__global__ __launch_bounds__(256)
void pgnn_edge_fused(const int*   __restrict__ src,
                     const int*   __restrict__ dst,
                     const float* __restrict__ sp_dist,
                     const int*   __restrict__ anchor_eid,
                     const float* __restrict__ u_feat,
                     const float* __restrict__ v_feat,
                     const float* __restrict__ W_out,
                     const float* __restrict__ b_out,
                     float* __restrict__ out_pos,   // [N, K]
                     float* __restrict__ out_str)   // [N, OUT_DIM]
{
    const int lane = threadIdx.x & 31;
    const int n    = blockIdx.x * 8 + (threadIdx.x >> 5);

    const float2 w  = ((const float2*)W_out)[lane];
    const float  b0 = b_out[0];

    float2 acc; acc.x = 0.f; acc.y = 0.f;
    const long base = (long)n * KANCH;

    for (int k = 0; k < KANCH; ++k) {
        const int   eid = anchor_eid[base + k];
        const int   s   = src[eid];
        const int   d   = dst[eid];
        const float sp  = sp_dist[eid];

        const float2 uf = ((const float2*)(u_feat + (long)s * OUT_DIM))[lane];
        const float2 vf = ((const float2*)(v_feat + (long)d * OUT_DIM))[lane];

        const float mx = fmaxf(fmaf(uf.x, sp, vf.x), 0.f);
        const float my = fmaxf(fmaf(uf.y, sp, vf.y), 0.f);
        acc.x += mx; acc.y += my;

        // dot(msg, W_out) reduced across the wave (wave32)
        float r = fmaf(mx, w.x, my * w.y);
        #pragma unroll
        for (int off = 16; off > 0; off >>= 1)
            r += __shfl_xor(r, off, 32);
        if (lane == 0)
            out_pos[base + k] = r + b0;
    }

    float2 m; m.x = acc.x * (1.0f / KANCH); m.y = acc.y * (1.0f / KANCH);
    ((float2*)(out_str + (long)n * OUT_DIM))[lane] = m;
}

// ---------------------------------------------------------------------------
extern "C" void kernel_launch(void* const* d_in, const int* in_sizes, int n_in,
                              void* d_out, int out_size, void* d_ws, size_t ws_size,
                              hipStream_t stream)
{
    const float* feature = (const float*)d_in[0];
    const int*   src     = (const int*)  d_in[1];
    const int*   dst     = (const int*)  d_in[2];
    const float* sp_dist = (const float*)d_in[3];
    const int*   aeid    = (const int*)  d_in[4];
    // d_in[5] = dists_max: dead in the reference (only its shape is used)
    const float* Wu      = (const float*)d_in[6];
    const float* bu      = (const float*)d_in[7];
    const float* Wv      = (const float*)d_in[8];
    const float* bv      = (const float*)d_in[9];
    const float* Wout    = (const float*)d_in[10];
    const float* bout    = (const float*)d_in[11];

    float* u_feat = (float*)d_ws;                    // [N, 64]
    float* v_feat = u_feat + (long)NNODES * OUT_DIM; // [N, 64]

    float* out_pos = (float*)d_out;                  // [N, K]
    float* out_str = out_pos + (long)NNODES * KANCH; // [N, 64]

    pgnn_proj_wmma<<<NBLK_A, 256, 0, stream>>>(
        feature, Wu, bu, Wv, bv, u_feat, v_feat);

    pgnn_edge_fused<<<NNODES / 8, 256, 0, stream>>>(
        src, dst, sp_dist, aeid, u_feat, v_feat, Wout, bout, out_pos, out_str);
}
